// SimpleConcatAttention_44401371906483
// MI455X (gfx1250) — matrix operand order repaired and checked by reading, not compiled, not run on hardware
//
#include <hip/hip_runtime.h>
#include <hip/hip_bf16.h>

// ---------------------------------------------------------------------------
// SimpleConcatAttention on MI455X (gfx1250): bf16 WMMA + TDM async staging.
//   query [512,1024] f32, key [16,2048,1024] f32, W [1024,1024] f32,
//   att_mask [16,512,2048] bool(1B)
//   d_out = out [16,512,1024] f32 ++ att [16,512,2048] f32
// Pipeline (all GEMM operands pre-converted to bf16 in ws; TDM stages tiles):
//   0) cvt query/W/key -> bf16 ws       (key bf16 = 67MB, L2-resident on 192MB L2)
//   1) qWb  = (qB @ WB) * (1/32)        -> ws bf16          (B K-major)
//   2) att  = qWb @ keyB^T, mask -inf   -> d_out[att] f32   (B N-major)
//   3) row softmax in place, also emit attB bf16 -> ws
//   4) out  = attB @ keyB               -> d_out[out] f32   (B K-major)
// ws need: (qB 0.5M + WB 1M + qWb 0.5M + keyB 32M + attB 16M) bf16 ~= 105 MB.
// ---------------------------------------------------------------------------

typedef __attribute__((ext_vector_type(16))) __bf16 v16bf;
typedef __attribute__((ext_vector_type(8)))  float  v8f;
typedef __attribute__((ext_vector_type(4)))  unsigned int v4u;
typedef __attribute__((ext_vector_type(8)))  int v8i;
typedef __attribute__((ext_vector_type(4)))  int v4i;

#define BM 64
#define BN 64
#define KS 32

#if defined(__gfx1250__) && __has_builtin(__builtin_amdgcn_tensor_load_to_lds)
#define HAVE_TDM 1
#else
#define HAVE_TDM 0
#endif

__device__ __forceinline__ unsigned short f2bf(float f) {
    union { float f; unsigned u; } x; x.f = f;
    unsigned r = x.u + 0x7FFFu + ((x.u >> 16) & 1u);   // RNE
    return (unsigned short)(r >> 16);
}

union Frag { v16bf v; unsigned u[8]; };

#if HAVE_TDM
// Issue one TDM 2D tile load: tile_h rows x tile_w bf16 elems, row stride in
// elems, raw-copied to contiguous LDS at lds_off. D# layout per ISA 8.3/8.4.
// This toolchain exposes the 6-arg builtin:
//   (uint32x4 g0, int32x8 g1, int32x4 g2, int32x4 g3, int32x8 extra, i32 cpol)
__device__ __forceinline__ void tdm_load_2d(unsigned lds_off, unsigned long long gaddr,
                                            unsigned tile_w, unsigned tile_h,
                                            unsigned long long stride_elems)
{
    v4u g0;
    g0[0] = 1u;                                               // count=1
    g0[1] = lds_off;                                          // LDS byte address
    g0[2] = (unsigned)(gaddr & 0xFFFFFFFFu);                  // global addr lo
    g0[3] = (unsigned)((gaddr >> 32) & 0x01FFFFFFu)           // global addr hi (57b)
            | (2u << 30);                                     // type=2 (image)
    const unsigned td0 = (unsigned)stride_elems;              // tensor_dim0 >= tile_w
    const unsigned td1 = 0x40000000u;                         // huge tensor_dim1
    v8i g1;
    g1[0] = 0x00010000;                                       // data_size=1 (2B), rest 0
    g1[1] = (int)((td0 & 0xFFFFu) << 16);                     // tensor_dim0[15:0]
    g1[2] = (int)((td0 >> 16) | ((td1 & 0xFFFFu) << 16));     // dim0 hi | dim1 lo
    g1[3] = (int)((td1 >> 16) | (tile_w << 16));              // dim1 hi | tile_dim0
    g1[4] = (int)(tile_h & 0xFFFFu);                          // tile_dim1, tile_dim2=0
    g1[5] = (int)(unsigned)(stride_elems & 0xFFFFFFFFu);      // dim0_stride lo32
    g1[6] = (int)(unsigned)((stride_elems >> 32) & 0xFFFFu);  // dim0_stride hi16
    g1[7] = 0;                                                // dim1_stride (unused)
    const v4i z4 = {0, 0, 0, 0};                              // 2D: groups 2/3 zero
    const v8i z8 = {0, 0, 0, 0, 0, 0, 0, 0};
    __builtin_amdgcn_tensor_load_to_lds(g0, g1, z4, z4, z8, 0);
}
#endif

// BKMAJOR=1: B stored [K][N]; BKMAJOR=0: B stored [N][K] (= B^T rows, contig K)
// EPI: 0 = scale + bf16 store, 1 = mask->-inf + f32 store, 2 = plain f32 store
template <int BKMAJOR, int EPI>
__global__ __launch_bounds__(256)
void gemm_bf16_wmma(const unsigned short* __restrict__ A, long strideA, int lda,
                    const unsigned short* __restrict__ B, long strideB, int ldb,
                    void* __restrict__ Cv, long strideC, int ldc,
                    const unsigned char* __restrict__ mask, long strideMask,
                    int N, int K, float scale)
{
    __shared__ __attribute__((aligned(16))) unsigned short At[2][BM * KS];
    __shared__ __attribute__((aligned(16))) unsigned short Bt[2][BM * KS];

    const int bz = blockIdx.z;
    const unsigned short* Ab = A + (long)bz * strideA;
    const unsigned short* Bb = B + (long)bz * strideB;

    const int rowBase = blockIdx.y * BM;
    const int colBase = blockIdx.x * BN;

    const int t = threadIdx.x, lane = t & 31, wave = t >> 5;
    const int wrow = wave >> 1;   // 0..3 : 16-row strip
    const int wcol = wave & 1;    // 0..1 : 32-col strip
    const int steps = K / KS;

    v8f acc0 = {}, acc1 = {};

#if HAVE_TDM
    auto issue = [&](int s) {
        const int k0 = s * KS;
        const int bs = s & 1;
        tdm_load_2d((unsigned)(unsigned long long)(const void*)&At[bs][0],
                    (unsigned long long)(const void*)(Ab + (long)rowBase * lda + k0),
                    KS, BM, (unsigned long long)lda);
        if (BKMAJOR)
            tdm_load_2d((unsigned)(unsigned long long)(const void*)&Bt[bs][0],
                        (unsigned long long)(const void*)(Bb + (long)k0 * ldb + colBase),
                        BN, KS, (unsigned long long)ldb);
        else
            tdm_load_2d((unsigned)(unsigned long long)(const void*)&Bt[bs][0],
                        (unsigned long long)(const void*)(Bb + (long)colBase * ldb + k0),
                        KS, BM, (unsigned long long)ldb);
    };
    if (wave == 0) issue(0);
#endif

    for (int s = 0; s < steps; ++s) {
        const int bs = s & 1;
#if HAVE_TDM
        if (wave == 0) __builtin_amdgcn_s_wait_tensorcnt(0);  // buf[s] landed
        __syncthreads();                                      // publish to all waves
        if (wave == 0 && s + 1 < steps) issue(s + 1);         // prefetch other buffer
#else
        {   // manual double-buffered staging: 16B per thread per tile
            const int k0 = s * KS;
            {
                const int r = t >> 2, ko = (t & 3) * 8;
                *(uint4*)&At[bs][r * KS + ko] =
                    *(const uint4*)(Ab + (long)(rowBase + r) * lda + k0 + ko);
            }
            if (BKMAJOR) {
                const int kr = t >> 3, no = (t & 7) * 8;
                *(uint4*)&Bt[bs][kr * BN + no] =
                    *(const uint4*)(Bb + (long)(k0 + kr) * ldb + colBase + no);
            } else {
                const int nr = t >> 2, ko = (t & 3) * 8;
                *(uint4*)&Bt[bs][nr * KS + ko] =
                    *(const uint4*)(Bb + (long)(colBase + nr) * ldb + k0 + ko);
            }
        }
        __syncthreads();
#endif
        // ---- A fragment (contiguous k = 16*(lane>>4) + s, same order as B) ----
        Frag af, b0, b1;
        {
            const int m  = wrow * 16 + (lane & 15);
            const int hb = (lane >> 4) * 16;
            const unsigned short* Ar = &At[bs][m * KS];
            #pragma unroll
            for (int i = 0; i < 4; ++i) {
                af.u[i]     = *(const unsigned*)&Ar[hb + 2 * i];
                af.u[4 + i] = *(const unsigned*)&Ar[8 + hb + 2 * i];
            }
        }
        // ---- B fragments (contiguous k order as A: half-lane = K block of 16) ----
        {
            const int n0 = wcol * 32 + (lane & 15);
            const int hb = (lane >> 4) * 16;
            if (BKMAJOR) {
                #pragma unroll
                for (int i = 0; i < 8; ++i) {
                    const int kk = hb + 2 * i;
                    b0.u[i] = (unsigned)Bt[bs][kk * BN + n0] |
                              ((unsigned)Bt[bs][(kk + 1) * BN + n0] << 16);
                    b1.u[i] = (unsigned)Bt[bs][kk * BN + n0 + 16] |
                              ((unsigned)Bt[bs][(kk + 1) * BN + n0 + 16] << 16);
                }
            } else {
                #pragma unroll
                for (int i = 0; i < 8; ++i) {
                    b0.u[i] = *(const unsigned*)&Bt[bs][n0 * KS + hb + 2 * i];
                    b1.u[i] = *(const unsigned*)&Bt[bs][(n0 + 16) * KS + hb + 2 * i];
                }
            }
        }
        acc0 = __builtin_amdgcn_wmma_f32_16x16x32_bf16(false, af.v, false, b0.v,
                                                       (short)0, acc0, false, false);
        acc1 = __builtin_amdgcn_wmma_f32_16x16x32_bf16(false, af.v, false, b1.v,
                                                       (short)0, acc1, false, false);
#if !HAVE_TDM
        __syncthreads();
#endif
    }

    // ---- epilogue (C/D layout: m = r + (lane>>4)*8, n = lane&15) ----
    const int mBase = rowBase + wrow * 16 + (lane >> 4) * 8;
    const int c0    = colBase + wcol * 32 + (lane & 15);
    #pragma unroll
    for (int r = 0; r < 8; ++r) {
        const int row = mBase + r;
        float v0 = acc0[r], v1 = acc1[r];
        if (EPI == 0) {
            unsigned short* Cb = (unsigned short*)Cv + (long)bz * strideC;
            Cb[(long)row * ldc + c0]      = f2bf(v0 * scale);
            Cb[(long)row * ldc + c0 + 16] = f2bf(v1 * scale);
        } else {
            float* Cb = (float*)Cv + (long)bz * strideC;
            if (EPI == 1) {
                const unsigned char* mk = mask + (long)bz * strideMask + (long)row * N;
                v0 = mk[c0]      ? v0 : -__builtin_inff();
                v1 = mk[c0 + 16] ? v1 : -__builtin_inff();
            }
            Cb[(long)row * ldc + c0]      = v0;
            Cb[(long)row * ldc + c0 + 16] = v1;
        }
    }
}

// fp32 -> bf16 bulk convert, 8 elements / thread
__global__ __launch_bounds__(256)
void cvt_f32_bf16(const float* __restrict__ s, unsigned short* __restrict__ d)
{
    const long i = ((long)blockIdx.x * 256 + threadIdx.x) * 8;
    float4 a = *(const float4*)(s + i);
    float4 b = *(const float4*)(s + i + 4);
    union { unsigned short us[8]; uint4 v; } o;
    o.us[0] = f2bf(a.x); o.us[1] = f2bf(a.y); o.us[2] = f2bf(a.z); o.us[3] = f2bf(a.w);
    o.us[4] = f2bf(b.x); o.us[5] = f2bf(b.y); o.us[6] = f2bf(b.z); o.us[7] = f2bf(b.w);
    *(uint4*)(d + i) = o.v;
}

// Row softmax over N=2048 (one 256-thread block per row); also emit bf16 copy.
__global__ __launch_bounds__(256)
void softmax_rows(float* __restrict__ att, unsigned short* __restrict__ attB)
{
    const long row = blockIdx.x;
    float* p = att + row * 2048;
    unsigned short* pb = attB + row * 2048;
    const int t = threadIdx.x;
    __shared__ float red[8];

    float v[8];
    float mx = -__builtin_inff();
    #pragma unroll
    for (int i = 0; i < 8; ++i) { v[i] = p[t + i * 256]; mx = fmaxf(mx, v[i]); }
    #pragma unroll
    for (int off = 16; off; off >>= 1) mx = fmaxf(mx, __shfl_xor(mx, off, 32));
    if ((t & 31) == 0) red[t >> 5] = mx;
    __syncthreads();
    mx = red[0];
    #pragma unroll
    for (int i = 1; i < 8; ++i) mx = fmaxf(mx, red[i]);

    float sum = 0.f;
    #pragma unroll
    for (int i = 0; i < 8; ++i) { v[i] = __expf(v[i] - mx); sum += v[i]; }
    #pragma unroll
    for (int off = 16; off; off >>= 1) sum += __shfl_xor(sum, off, 32);
    __syncthreads();
    if ((t & 31) == 0) red[t >> 5] = sum;
    __syncthreads();
    sum = 0.f;
    #pragma unroll
    for (int i = 0; i < 8; ++i) sum += red[i];

    const float inv = 1.0f / sum;
    #pragma unroll
    for (int i = 0; i < 8; ++i) {
        const float o = v[i] * inv;
        p[t + i * 256]  = o;
        pb[t + i * 256] = f2bf(o);
    }
}

extern "C" void kernel_launch(void* const* d_in, const int* in_sizes, int n_in,
                              void* d_out, int out_size, void* d_ws, size_t ws_size,
                              hipStream_t stream)
{
    (void)in_sizes; (void)n_in; (void)out_size; (void)ws_size;
    const long B = 16, Kq = 512, N = 2048, D = 1024;

    const float*         query = (const float*)d_in[0];
    const float*         key   = (const float*)d_in[1];
    const float*         W     = (const float*)d_in[2];
    const unsigned char* mask  = (const unsigned char*)d_in[3];

    float* out = (float*)d_out;                 // [B,Kq,D]
    float* att = out + B * Kq * D;              // [B,Kq,N]

    unsigned short* qB   = (unsigned short*)d_ws;      // [Kq,D]
    unsigned short* WB   = qB   + Kq * D;              // [D,D]
    unsigned short* qWb  = WB   + D * D;               // [Kq,D]
    unsigned short* keyB = qWb  + Kq * D;              // [B,N,D]
    unsigned short* attB = keyB + B * N * D;           // [B,Kq,N]

    dim3 blk(256);

    // 0) bulk fp32 -> bf16 conversions (keyB stays hot in 192MB L2 afterwards)
    cvt_f32_bf16<<<dim3((unsigned)(Kq * D / 2048)), blk, 0, stream>>>(query, qB);
    cvt_f32_bf16<<<dim3((unsigned)(D * D / 2048)), blk, 0, stream>>>(W, WB);
    cvt_f32_bf16<<<dim3((unsigned)(B * N * D / 2048)), blk, 0, stream>>>(key, keyB);

    // 1) qWb = (qB @ WB) / 32          M=512 N=1024 K=1024, B K-major, bf16 out
    gemm_bf16_wmma<1, 0><<<dim3(D / BN, Kq / BM, 1), blk, 0, stream>>>(
        qB, 0, (int)D,  WB, 0, (int)D,  qWb, 0, (int)D,
        nullptr, 0,  (int)D, (int)D, 0.03125f);

    // 2) att = qWb @ keyB[b]^T masked  M=512 N=2048 K=1024, B N-major (key rows)
    gemm_bf16_wmma<0, 1><<<dim3(N / BN, Kq / BM, B), blk, 0, stream>>>(
        qWb, 0, (int)D,  keyB, N * D, (int)D,  att, Kq * N, (int)N,
        mask, Kq * N,  (int)N, (int)D, 1.0f);

    // 3) softmax rows in place + bf16 copy
    softmax_rows<<<dim3((unsigned)(B * Kq)), blk, 0, stream>>>(att, attB);

    // 4) out = attB @ keyB[b]          M=512 N=1024 K=2048, B K-major
    gemm_bf16_wmma<1, 2><<<dim3(D / BN, Kq / BM, B), blk, 0, stream>>>(
        attB, Kq * N, (int)N,  keyB, N * D, (int)D,  out, Kq * D, (int)D,
        nullptr, 0,  (int)D, (int)N, 1.0f);
}
